// xNon_local_30210799960322
// MI455X (gfx1250) — matrix operand order repaired and checked by reading, not compiled, not run on hardware
//
#include <hip/hip_runtime.h>
#include <hip/hip_bf16.h>

typedef __attribute__((ext_vector_type(16))) __bf16 v16bf;
typedef __attribute__((ext_vector_type(8)))  __bf16 v8bf;
typedef __attribute__((ext_vector_type(8)))  float  v8f;

#define HW_TOT 65536
#define CCH    64

// ---------------- workspace layout ----------------
constexpr int OFF_S    = 0;       // [4][64][64] Gram accumulators
constexpr int OFF_SM   = 16384;   // [4][64] channel sums of m
constexpr int OFF_SX   = 16640;   // [4][64] channel sums of x
constexpr int OFF_AF   = 16896;   // A = Wcat[:, :32] @ Wm   [64][64]
constexpr int OFF_BF   = 20992;   // B = Wcat[:, 32:] @ Wx   [64][64]
constexpr int OFF_B0   = 25088;   // folded bias [64]
constexpr int OFF_BMIX = 25152;   // w1*bq + (1-w1)*bsig [64]
constexpr int OFF_WMIX = 25216;   // w1*Wq + (1-w1)*Wsig [64][64]
constexpr int OFF_CV   = 29312;   // [4][64] output constant
constexpr size_t BF_BYTE_OFF = 29568u * 4u;   // 128B aligned
constexpr int BF_A  = 0;          // A in WMMA-A layout   [4][2][32][16]
constexpr int BF_B  = 4096;       // B in WMMA-A layout
constexpr int BF_M1 = 8192;       // per batch M1 [4][4096]
constexpr int BF_M2 = 24576;      // per batch M2 [4][4096]

// WMMA bf16 fragment maps (wave32), per CDNA5 ISA 7.12.2:
//  A 16x32 : lane l -> M=l&15 ; elem i -> K = (i>>3)*16 + ((l&16)?8:0) + (i&7)
//  B 32x16 : lane l -> N=l&15 ; elem i -> K = (i>>3)*16 + ((l&16)?8:0) + (i&7)
//  C/D f32 : lane l -> N=l&15 ; elem i -> M = ((l&16)?8:0) + i

// ---------------- kernel 1: fold weights, zero accumulators ----------------
__global__ void __launch_bounds__(256)
prep_kernel(const float* Wx, const float* bx, const float* Wm, const float* bm,
            const float* Wcat, const float* bcat,
            const float* Wq, const float* Wsig, const float* bq, const float* bsig,
            const float* w1p, float* ws, __bf16* wsbf) {
  __shared__ float lA[4096];
  __shared__ float lB[4096];
  const int t = threadIdx.x;
  for (int idx = t; idx < 16896; idx += 256) ws[idx] = 0.f;   // zero S, sm, sx
  const float w1 = w1p[0];
  for (int s = t; s < 4096; s += 256) {
    int o = s >> 6, c = s & 63;
    float fa = 0.f, fb = 0.f;
    for (int j = 0; j < 32; ++j) {
      fa += Wcat[o * 64 + j]      * Wm[j * 64 + c];
      fb += Wcat[o * 64 + 32 + j] * Wx[j * 64 + c];
    }
    lA[s] = fa; lB[s] = fb;
    ws[OFF_AF + s] = fa;
    ws[OFF_BF + s] = fb;
    ws[OFF_WMIX + s] = w1 * Wq[s] + (1.f - w1) * Wsig[s];
  }
  if (t < 64) {
    float acc = bcat[t];
    for (int j = 0; j < 32; ++j)
      acc += Wcat[t * 64 + j] * bm[j] + Wcat[t * 64 + 32 + j] * bx[j];
    ws[OFF_B0 + t] = acc;
    ws[OFF_BMIX + t] = w1 * bq[t] + (1.f - w1) * bsig[t];
  }
  __syncthreads();
  for (int s = t; s < 4096; s += 256) {   // bf16 WMMA-A layout copies
    int i = s & 15, lane = (s >> 4) & 31, kk = (s >> 9) & 1, mb = s >> 10;
    int M = mb * 16 + (lane & 15);
    int K = kk * 32 + (i >> 3) * 16 + ((lane & 16) ? 8 : 0) + (i & 7);
    wsbf[BF_A + s] = (__bf16)lA[M * 64 + K];
    wsbf[BF_B + s] = (__bf16)lB[M * 64 + K];
  }
}

// ---------------- kernel 2: accumulate S = xm xm^T and channel sums ----------------
// 64-pixel chunks; weights live in LDS so steady-state VGPR use is small.
__global__ void __launch_bounds__(256, 4)
stat_kernel(const float* __restrict__ x, const float* __restrict__ m,
            float* ws, const __bf16* __restrict__ wsbf) {
  __shared__ __align__(32) float lds_raw[12288];        // 48 KB
  __bf16* ldsW   = (__bf16*)lds_raw;                    // [8192] A|B weight frags
  __bf16* ldsBm  = ldsW + 8192;                         // [2][4][32][16] m tile
  __bf16* ldsBx  = ldsW + 12288;                        // x tile
  __bf16* ldsAxm = ldsW + 16384;                        // [4][2][32][16] xm A-frag
  __bf16* ldsBxm = ldsW + 20480;                        // [2][4][32][16] xm^T B-frag

  const int n = blockIdx.y;
  const int t = threadIdx.x;
  const int lane = t & 31;
  const int w = t >> 5;
  const int basech = (lane & 16) ? 8 : 0;
  const size_t base_in = (size_t)n * CCH * HW_TOT;
  const float* b0v = ws + OFF_B0;

  // weights (A|B contiguous, 8192 bf16) -> LDS
#pragma unroll
  for (int r = 0; r < 2; ++r) {
    int idx = r * 256 + t;
    *(v16bf*)&ldsW[idx * 16] = *(const v16bf*)&wsbf[BF_A + idx * 16];
  }
  // per-wave bias vectors (tiles T = w, w+8 -> fixed mb pair)
  v8f bias[2];
#pragma unroll
  for (int tt = 0; tt < 2; ++tt) {
    int mb = (w + tt * 8) >> 2;
#pragma unroll
    for (int i = 0; i < 8; ++i) bias[tt][i] = b0v[mb * 16 + basech + i];
  }

  // staging assignment: one group of 8 channels x 2 pixels per thread
  const int plp = (lane & 15) * 2 + ((t & 16) ? 2 : 0) * 0 + ((t >> 0) & 0);  // placeholder
  (void)plp;
  const int gpl = (t & 31) * 2;          // even pixel 0..62
  const int c8  = t >> 5;                // channel octet 0..7
  const int skk   = c8 >> 2;
  const int shalf = (c8 & 1) ? 16 : 0;
  const int si0   = ((c8 & 3) >> 1) * 8;
  const int s0 = (((skk * 4 + (gpl >> 4)) * 32) + shalf + (gpl & 15)) * 16 + si0;

  float partM[8], partX[8];
#pragma unroll
  for (int j = 0; j < 8; ++j) { partM[j] = 0.f; partX[j] = 0.f; }

  v8f accS[2]; accS[0] = (v8f){}; accS[1] = (v8f){};

  for (int chunk = blockIdx.x; chunk < HW_TOT / 64; chunk += gridDim.x) {
    const int p0 = chunk * 64;
    __syncthreads();
    {
      v8bf pm0, pm1, px0, px1;
#pragma unroll
      for (int j = 0; j < 8; ++j) {
        const size_t off = base_in + (size_t)(c8 * 8 + j) * HW_TOT + p0 + gpl;
        float2 vm = *(const float2*)(m + off);
        float2 vx = *(const float2*)(x + off);
        partM[j] += vm.x + vm.y;
        partX[j] += vx.x + vx.y;
        pm0[j] = (__bf16)vm.x; pm1[j] = (__bf16)vm.y;
        px0[j] = (__bf16)vx.x; px1[j] = (__bf16)vx.y;
      }
      *(v8bf*)&ldsBm[s0]      = pm0;
      *(v8bf*)&ldsBm[s0 + 16] = pm1;   // next pixel = next lane = +16 elems
      *(v8bf*)&ldsBx[s0]      = px0;
      *(v8bf*)&ldsBx[s0 + 16] = px1;
    }
    __syncthreads();
    // xm = A*m + B*x + b0 ; wave w owns tiles T = w, w+8
#pragma unroll
    for (int tt = 0; tt < 2; ++tt) {
      const int T = w + tt * 8;
      const int mb = T >> 2, nb = T & 3;
      v8f acc = bias[tt];
#pragma unroll
      for (int kk = 0; kk < 2; ++kk) {
        v16bf am = *(const v16bf*)&ldsW[((mb * 2 + kk) * 32 + lane) * 16];
        v16bf dm = *(const v16bf*)&ldsBm[((kk * 4 + nb) * 32 + lane) * 16];
        acc = __builtin_amdgcn_wmma_f32_16x16x32_bf16(false, am, false, dm, (short)0, acc, false, false);
        v16bf ax = *(const v16bf*)&ldsW[4096 + ((mb * 2 + kk) * 32 + lane) * 16];
        v16bf dx = *(const v16bf*)&ldsBx[((kk * 4 + nb) * 32 + lane) * 16];
        acc = __builtin_amdgcn_wmma_f32_16x16x32_bf16(false, ax, false, dx, (short)0, acc, false, false);
      }
      // scatter xm into A-frag (rows) and B-frag (transposed) layouts
#pragma unroll
      for (int i = 0; i < 8; ++i) {
        int ch_m = basech + i;                  // ch % 16
        int pxl  = nb * 16 + (lane & 15);       // 0..63
        int kk2 = pxl >> 5, kl = pxl & 31;
        int low = kl & 15, hi = kl >> 4;
        __bf16 v = (__bf16)acc[i];
        int laneA = (low < 8) ? ch_m : (16 + ch_m);
        ldsAxm[((mb * 2 + kk2) * 32 + laneA) * 16 + (hi * 8 + (low & 7))] = v;
        int laneB = ((kl & 8) ? 16 : 0) + ch_m;
        ldsBxm[((kk2 * 4 + mb) * 32 + laneB) * 16 + ((kl >> 4) * 8 + (kl & 7))] = v;
      }
    }
    __syncthreads();
    // S += xm * xm^T ; wave w owns Gram tiles w and w+8, K=64 pixels
#pragma unroll
    for (int tt = 0; tt < 2; ++tt) {
      const int T = w + tt * 8;
      const int mbS = T >> 2, nbS = T & 3;
#pragma unroll
      for (int kk2 = 0; kk2 < 2; ++kk2) {
        v16bf a = *(const v16bf*)&ldsAxm[((mbS * 2 + kk2) * 32 + lane) * 16];
        v16bf f = *(const v16bf*)&ldsBxm[((kk2 * 4 + nbS) * 32 + lane) * 16];
        accS[tt] = __builtin_amdgcn_wmma_f32_16x16x32_bf16(false, a, false, f, (short)0, accS[tt], false, false);
      }
    }
  }
  // flush Gram partials
  {
    float* Sg = ws + OFF_S + n * 4096;
#pragma unroll
    for (int tt = 0; tt < 2; ++tt) {
      const int T = w + tt * 8;
      const int mb = T >> 2, nb = T & 3;
#pragma unroll
      for (int i = 0; i < 8; ++i)
        atomicAdd(&Sg[(mb * 16 + basech + i) * 64 + nb * 16 + (lane & 15)], accS[tt][i]);
    }
  }
  // reduce channel sums: partial j of thread t covers ch = (t>>5)*8 + j
  __syncthreads();
#pragma unroll
  for (int j = 0; j < 8; ++j) {
    lds_raw[t * 16 + j]     = partM[j];
    lds_raw[t * 16 + 8 + j] = partX[j];
  }
  __syncthreads();
  if (t < 128) {
    int ch = t & 63, isX = t >> 6;
    int oct = ch >> 3;
    int slot = (ch & 7) + (isX ? 8 : 0);
    float s = 0.f;
    for (int k = 0; k < 32; ++k) s += lds_raw[(oct * 32 + k) * 16 + slot];
    atomicAdd(&ws[(isX ? OFF_SX : OFF_SM) + n * 64 + ch], s);
  }
}

// ---------------- kernel 3: softmax core + fold into M1, M2, cvec ----------------
__global__ void __launch_bounds__(1024)
attn_kernel(const float* Wk, const float* bk, const float* Wv, const float* bv,
            const float* Wout, const float* bout, float* ws, __bf16* wsbf) {
  __shared__ float b1[4096];
  __shared__ float b2[4096];
  __shared__ float b3[4096];
  __shared__ float svec[64], uvec[64], vvec[64], pbv[64], wpb[64];
  const int n = blockIdx.x;
  const int t = threadIdx.x;
  const float* Af   = ws + OFF_AF;
  const float* Bf   = ws + OFF_BF;
  const float* b0v  = ws + OFF_B0;
  const float* Wmix = ws + OFF_WMIX;
  const float* bmix = ws + OFF_BMIX;
  const float* smv  = ws + OFF_SM + n * 64;
  const float* sxv  = ws + OFF_SX + n * 64;
  const float* Sg   = ws + OFF_S + n * 4096;

  for (int s = t; s < 4096; s += 1024) b1[s] = Sg[s];         // b1 = S
  if (t < 64) {                                               // s = A sm + B sx + HW b0
    float a = (float)HW_TOT * b0v[t];
    for (int c = 0; c < 64; ++c) a += Af[t * 64 + c] * smv[c] + Bf[t * 64 + c] * sxv[c];
    svec[t] = a;
  }
  __syncthreads();
  if (t < 64) {                                               // u = Wk s
    float a = 0.f;
    for (int c = 0; c < 64; ++c) a += Wk[t * 64 + c] * svec[c];
    uvec[t] = a;
  } else if (t < 128) {                                       // v = Wmix s
    int j = t - 64; float a = 0.f;
    for (int c = 0; c < 64; ++c) a += Wmix[j * 64 + c] * svec[c];
    vvec[j] = a;
  }
  __syncthreads();
  for (int s = t; s < 4096; s += 1024) {                      // b2 = Wk S
    int i = s >> 6, j = s & 63; float a = 0.f;
    for (int c = 0; c < 64; ++c) a += Wk[i * 64 + c] * b1[c * 64 + j];
    b2[s] = a;
  }
  __syncthreads();
  for (int s = t; s < 4096; s += 1024) {                      // b1 = pw
    int i = s >> 6, j = s & 63;
    float a = uvec[i] * bmix[j] + bk[i] * vvec[j] + (float)HW_TOT * bk[i] * bmix[j];
    for (int c = 0; c < 64; ++c) a += b2[i * 64 + c] * Wmix[j * 64 + c];
    b1[s] = a;
  }
  __syncthreads();
  if (t < 64) {                                               // softmax rows -> P
    float mx = -1e30f;
    for (int j = 0; j < 64; ++j) mx = fmaxf(mx, b1[t * 64 + j]);
    float sum = 0.f;
    for (int j = 0; j < 64; ++j) { float e = __expf(b1[t * 64 + j] - mx); b1[t * 64 + j] = e; sum += e; }
    float inv = 1.f / sum;
    for (int j = 0; j < 64; ++j) b1[t * 64 + j] *= inv;
  }
  __syncthreads();
  for (int s = t; s < 4096; s += 1024) {                      // b2 = P Wv
    int i = s >> 6, c = s & 63; float a = 0.f;
    for (int j = 0; j < 64; ++j) a += b1[i * 64 + j] * Wv[j * 64 + c];
    b2[s] = a;
  }
  if (t < 64) {                                               // pbv = P bv
    float a = 0.f;
    for (int j = 0; j < 64; ++j) a += b1[t * 64 + j] * bv[j];
    pbv[t] = a;
  }
  __syncthreads();
  for (int s = t; s < 4096; s += 1024) {                      // b3 = G = Wout P Wv
    int o = s >> 6, c = s & 63; float a = 0.f;
    for (int j = 0; j < 64; ++j) a += Wout[o * 64 + j] * b2[j * 64 + c];
    b3[s] = a;
  }
  if (t < 64) {                                               // wpb = Wout pbv + bout
    float a = bout[t];
    for (int j = 0; j < 64; ++j) a += Wout[t * 64 + j] * pbv[j];
    wpb[t] = a;
  }
  __syncthreads();
  for (int s = t; s < 4096; s += 1024) {                      // M1 = G A ; M2 = G B
    int o = s >> 6, c = s & 63;
    float a1 = 0.f, a2 = 0.f;
    for (int k = 0; k < 64; ++k) {
      float g = b3[o * 64 + k];
      a1 += g * Af[k * 64 + c];
      a2 += g * Bf[k * 64 + c];
    }
    b1[s] = a1; b2[s] = a2;
  }
  if (t < 64) {                                               // cvec = G b0 + wpb
    float a = wpb[t];
    for (int k = 0; k < 64; ++k) a += b3[t * 64 + k] * b0v[k];
    ws[OFF_CV + n * 64 + t] = a;
  }
  __syncthreads();
  __bf16* M1bf = wsbf + BF_M1 + n * 4096;
  __bf16* M2bf = wsbf + BF_M2 + n * 4096;
  for (int s = t; s < 4096; s += 1024) {
    int i = s & 15, lane = (s >> 4) & 31, kk = (s >> 9) & 1, mb = s >> 10;
    int M = mb * 16 + (lane & 15);
    int K = kk * 32 + (i >> 3) * 16 + ((lane & 16) ? 8 : 0) + (i & 7);
    M1bf[s] = (__bf16)b1[M * 64 + K];
    M2bf[s] = (__bf16)b2[M * 64 + K];
  }
}

// ---------------- kernel 4: out = M1 m + M2 x + cvec + x ----------------
__global__ void __launch_bounds__(256, 4)
out_kernel(const float* __restrict__ x, const float* __restrict__ m,
           const float* __restrict__ ws, const __bf16* __restrict__ wsbf,
           float* __restrict__ out) {
  __shared__ __align__(32) float k4_raw[12288];   // 48 KB
  __bf16* ldsW   = (__bf16*)k4_raw;               // [8192] M1|M2 frags
  __bf16* ldsBm  = ldsW + 8192;                   // [2][4][32][16]
  __bf16* ldsBx  = ldsW + 12288;
  float*  lds_xf = &k4_raw[8192];                 // [64px][64ch] f32 x (exact residual)

  const int n = blockIdx.y;
  const int t = threadIdx.x;
  const int lane = t & 31;
  const int w = t >> 5;
  const int basech = (lane & 16) ? 8 : 0;
  const size_t base = (size_t)n * CCH * HW_TOT;
  const float* cvec = ws + OFF_CV + n * 64;
  const __bf16* M1bf = wsbf + BF_M1 + n * 4096;
  const __bf16* M2bf = wsbf + BF_M2 + n * 4096;

  // per-batch weight fragments -> LDS
  *(v16bf*)&ldsW[t * 16]        = *(const v16bf*)&M1bf[t * 16];
  *(v16bf*)&ldsW[4096 + t * 16] = *(const v16bf*)&M2bf[t * 16];
  // per-wave bias
  v8f bias[2];
#pragma unroll
  for (int tt = 0; tt < 2; ++tt) {
    int mb = (w + tt * 8) >> 2;
#pragma unroll
    for (int i = 0; i < 8; ++i) bias[tt][i] = cvec[mb * 16 + basech + i];
  }

  const int gpl = (t & 31) * 2;
  const int c8  = t >> 5;
  const int skk   = c8 >> 2;
  const int shalf = (c8 & 1) ? 16 : 0;
  const int si0   = ((c8 & 3) >> 1) * 8;
  const int s0 = (((skk * 4 + (gpl >> 4)) * 32) + shalf + (gpl & 15)) * 16 + si0;

  for (int chunk = blockIdx.x; chunk < HW_TOT / 64; chunk += gridDim.x) {
    const int p0 = chunk * 64;
    __syncthreads();
    {
      v8bf pm0, pm1, px0, px1;
      v8f  fx0, fx1;
#pragma unroll
      for (int j = 0; j < 8; ++j) {
        const size_t off = base + (size_t)(c8 * 8 + j) * HW_TOT + p0 + gpl;
        float2 vm = *(const float2*)(m + off);
        float2 vx = *(const float2*)(x + off);
        pm0[j] = (__bf16)vm.x; pm1[j] = (__bf16)vm.y;
        px0[j] = (__bf16)vx.x; px1[j] = (__bf16)vx.y;
        fx0[j] = vx.x; fx1[j] = vx.y;
      }
      *(v8bf*)&ldsBm[s0]      = pm0;
      *(v8bf*)&ldsBm[s0 + 16] = pm1;
      *(v8bf*)&ldsBx[s0]      = px0;
      *(v8bf*)&ldsBx[s0 + 16] = px1;
      *(v8f*)&lds_xf[gpl * 64 + c8 * 8]       = fx0;
      *(v8f*)&lds_xf[(gpl + 1) * 64 + c8 * 8] = fx1;
    }
    __syncthreads();
#pragma unroll
    for (int tt = 0; tt < 2; ++tt) {
      const int T = w + tt * 8;
      const int mb = T >> 2, nb = T & 3;
      v8f acc = bias[tt];
#pragma unroll
      for (int kk = 0; kk < 2; ++kk) {
        v16bf a1 = *(const v16bf*)&ldsW[((mb * 2 + kk) * 32 + lane) * 16];
        v16bf dm = *(const v16bf*)&ldsBm[((kk * 4 + nb) * 32 + lane) * 16];
        acc = __builtin_amdgcn_wmma_f32_16x16x32_bf16(false, a1, false, dm, (short)0, acc, false, false);
        v16bf a2 = *(const v16bf*)&ldsW[4096 + ((mb * 2 + kk) * 32 + lane) * 16];
        v16bf dx = *(const v16bf*)&ldsBx[((kk * 4 + nb) * 32 + lane) * 16];
        acc = __builtin_amdgcn_wmma_f32_16x16x32_bf16(false, a2, false, dx, (short)0, acc, false, false);
      }
      const int pxl = nb * 16 + (lane & 15);
      v8f xv = *(const v8f*)&lds_xf[pxl * 64 + mb * 16 + basech];
#pragma unroll
      for (int i = 0; i < 8; ++i) {
        int ch = mb * 16 + basech + i;
        __builtin_nontemporal_store(acc[i] + xv[i],
                                    &out[base + (size_t)ch * HW_TOT + p0 + pxl]);
      }
    }
  }
}

// ---------------- launch ----------------
extern "C" void kernel_launch(void* const* d_in, const int* in_sizes, int n_in,
                              void* d_out, int out_size, void* d_ws, size_t ws_size,
                              hipStream_t stream) {
  (void)in_sizes; (void)n_in; (void)out_size; (void)ws_size;
  const float* x    = (const float*)d_in[0];
  const float* m    = (const float*)d_in[1];
  const float* Wx   = (const float*)d_in[2];
  const float* bx   = (const float*)d_in[3];
  const float* Wm   = (const float*)d_in[4];
  const float* bm   = (const float*)d_in[5];
  const float* Wcat = (const float*)d_in[6];
  const float* bcat = (const float*)d_in[7];
  const float* Wq   = (const float*)d_in[8];
  const float* bq   = (const float*)d_in[9];
  const float* Wk   = (const float*)d_in[10];
  const float* bk   = (const float*)d_in[11];
  const float* Wv   = (const float*)d_in[12];
  const float* bv   = (const float*)d_in[13];
  const float* Wsig = (const float*)d_in[14];
  const float* bsig = (const float*)d_in[15];
  const float* Wout = (const float*)d_in[16];
  const float* bout = (const float*)d_in[17];
  const float* w1   = (const float*)d_in[18];
  float*  out  = (float*)d_out;
  float*  ws   = (float*)d_ws;
  __bf16* wsbf = (__bf16*)((char*)d_ws + BF_BYTE_OFF);

  prep_kernel<<<1, 256, 0, stream>>>(Wx, bx, Wm, bm, Wcat, bcat, Wq, Wsig, bq, bsig, w1, ws, wsbf);
  stat_kernel<<<dim3(64, 4), 256, 0, stream>>>(x, m, ws, wsbf);
  attn_kernel<<<dim3(4), 1024, 0, stream>>>(Wk, bk, Wv, bv, Wout, bout, ws, wsbf);
  out_kernel<<<dim3(128, 4), 256, 0, stream>>>(x, m, ws, wsbf, out);
}
